// MAMIL_56006373540263
// MI455X (gfx1250) — matrix operand order repaired and checked by reading, not compiled, not run on hardware
//
#include <hip/hip_runtime.h>
#include <hip/hip_bf16.h>

typedef __attribute__((ext_vector_type(16))) __bf16 bf16x16;
typedef __attribute__((ext_vector_type(8)))  __bf16 bf16x8;
typedef __attribute__((ext_vector_type(8)))  float  f32x8;

#define F32X8_ZERO {0.f,0.f,0.f,0.f,0.f,0.f,0.f,0.f}

// ---- WMMA fragment helpers (CDNA5 bf16 16x16x32 layouts) ----

__device__ inline f32x8 wmma_bf16(bf16x16 a, bf16x16 b, f32x8 c) {
  return __builtin_amdgcn_wmma_f32_16x16x32_bf16(
      /*neg_a=*/false, a, /*neg_b=*/false, b,
      /*c_mod=*/(short)0, c, /*reuse_a=*/false, /*reuse_b=*/false);
}

// A fragment from bf16 row-major [rows, ld]; lane holds row, K-chunks {kb+hf*8..+7, +16..+23}
__device__ inline bf16x16 load_a_bf16(const __bf16* base, int ld, int row, int kb, int hf) {
  const __bf16* p = base + (size_t)row * ld + kb + hf * 8;
  bf16x8 c0 = *(const bf16x8*)(p);
  bf16x8 c1 = *(const bf16x8*)(p + 16);
  bf16x16 r;
#pragma unroll
  for (int i = 0; i < 8; ++i) { r[i] = c0[i]; r[i + 8] = c1[i]; }
  return r;
}

// A fragment from f32 row-major source, converting to bf16 in-register
__device__ inline bf16x16 load_a_f32(const float* base, int ld, int row, int kb, int hf) {
  const float* p = base + (size_t)row * ld + kb + hf * 8;
  f32x8 c0 = *(const f32x8*)(p);
  f32x8 c1 = *(const f32x8*)(p + 16);
  bf16x16 r;
#pragma unroll
  for (int i = 0; i < 8; ++i) { r[i] = (__bf16)c0[i]; r[i + 8] = (__bf16)c1[i]; }
  return r;
}

// B fragment from pre-transposed bf16 weight Wt [Ncols, K]: lane holds col, K-chunks {kb+hf*8..+7, +16..+23}
__device__ inline bf16x16 load_b_bf16(const __bf16* wt, int ld, int col, int kb, int hf) {
  const __bf16* p = wt + (size_t)col * ld + kb + hf * 8;
  bf16x8 c0 = *(const bf16x8*)(p);
  bf16x8 c1 = *(const bf16x8*)(p + 16);
  bf16x16 r;
#pragma unroll
  for (int i = 0; i < 8; ++i) { r[i] = c0[i]; r[i + 8] = c1[i]; }
  return r;
}

// ---- k0: transpose + convert weights to bf16 ----
__global__ void k0_convert(const float* __restrict__ W1, const float* __restrict__ Wn,
                           const float* __restrict__ Wp, __bf16* __restrict__ W1t,
                           __bf16* __restrict__ Wnt, __bf16* __restrict__ Wpt) {
  int i = blockIdx.x * 256 + threadIdx.x;
  if (i < 512 * 128) { int k = i / 128, c = i % 128; W1t[(size_t)c * 512 + k] = (__bf16)W1[i]; }
  if (i < 128 * 128) { int k = i / 128, c = i % 128; Wnt[(size_t)c * 128 + k] = (__bf16)Wn[i]; }
  if (i < 256 * 256) { int k = i / 256, c = i % 256; Wpt[(size_t)c * 256 + k] = (__bf16)Wp[i]; }
}

// ---- k1: H = relu(x @ W1 + b1) -> Hbf [N,128] bf16 ----
__global__ void k1_h(const float* __restrict__ x, const __bf16* __restrict__ W1t,
                     const float* __restrict__ b1, __bf16* __restrict__ Hbf,
                     int N, int ntiles) {
  int w = threadIdx.x >> 5, lane = threadIdx.x & 31;
  int hf = lane >> 4, lr = lane & 15;
  int tile = blockIdx.x * 8 + w; if (tile >= ntiles) tile = ntiles - 1;
  int row = tile * 16 + lr; if (row >= N) row = N - 1;
  f32x8 acc[8];
#pragma unroll
  for (int i = 0; i < 8; ++i) acc[i] = (f32x8)F32X8_ZERO;
  for (int kb = 0; kb < 512; kb += 32) {
    bf16x16 a = load_a_f32(x, 512, row, kb, hf);
    bf16x16 bfr[8];
#pragma unroll
    for (int ct = 0; ct < 8; ++ct) bfr[ct] = load_b_bf16(W1t, 512, ct * 16 + lr, kb, hf);
#pragma unroll
    for (int ct = 0; ct < 8; ++ct) acc[ct] = wmma_bf16(a, bfr[ct], acc[ct]);
  }
#pragma unroll
  for (int ct = 0; ct < 8; ++ct) {
    int col = ct * 16 + lr;
    float bias = b1[col];
#pragma unroll
    for (int j = 0; j < 8; ++j) {
      int r = tile * 16 + j + hf * 8;
      if (r < N) {
        float v = fmaxf(acc[ct][j] + bias, 0.0f);
        Hbf[(size_t)r * 128 + col] = (__bf16)v;
      }
    }
  }
}

// ---- k2a: neighbor attention -> Hcbf [N,256] = [H | emb] ----
__global__ void k2a_attn(const __bf16* __restrict__ Hbf, const __bf16* __restrict__ Wnt,
                         const float* __restrict__ bn, __bf16* __restrict__ Hcbf,
                         int N, int ntiles) {
  __shared__ float sred[8 * 2 * 16 * 16];
  __shared__ float scs[8 * 32];
  __shared__ float al[8 * 16];
  int w = threadIdx.x >> 5, lane = threadIdx.x & 31;
  int hf = lane >> 4, lr = lane & 15;
  int tile = blockIdx.x * 8 + w; if (tile >= ntiles) tile = ntiles - 1;
  int i = tile * 16 + lr; if (i >= N) i = N - 1;
  int prev = (i == 0) ? 1 : i - 1;
  int nxt  = (i == N - 1) ? N - 2 : i + 1;
  f32x8 accP[8], accN[8];
#pragma unroll
  for (int q = 0; q < 8; ++q) { accP[q] = (f32x8)F32X8_ZERO; accN[q] = (f32x8)F32X8_ZERO; }
  for (int kb = 0; kb < 128; kb += 32) {
    bf16x16 aP = load_a_bf16(Hbf, 128, prev, kb, hf);
    bf16x16 aN = load_a_bf16(Hbf, 128, nxt,  kb, hf);
    bf16x16 bfr[8];
#pragma unroll
    for (int ct = 0; ct < 8; ++ct) bfr[ct] = load_b_bf16(Wnt, 128, ct * 16 + lr, kb, hf);
#pragma unroll
    for (int ct = 0; ct < 8; ++ct) {
      accP[ct] = wmma_bf16(aP, bfr[ct], accP[ct]);
      accN[ct] = wmma_bf16(aN, bfr[ct], accN[ct]);
    }
  }
  float sP[8], sN[8];
#pragma unroll
  for (int j = 0; j < 8; ++j) { sP[j] = 0.f; sN[j] = 0.f; }
#pragma unroll
  for (int ct = 0; ct < 8; ++ct) {
    int col = ct * 16 + lr;
    float bb = bn[col];
#pragma unroll
    for (int j = 0; j < 8; ++j) {
      int rg = tile * 16 + j + hf * 8; if (rg >= N) rg = N - 1;
      float h = (float)Hbf[(size_t)rg * 128 + col];
      sP[j] += tanhf(accP[ct][j] + bb) * h;
      sN[j] += tanhf(accN[ct][j] + bb) * h;
    }
  }
#pragma unroll
  for (int j = 0; j < 8; ++j) {
    int M = j + hf * 8;
    sred[((w * 2 + 0) * 16 + M) * 16 + lr] = sP[j];
    sred[((w * 2 + 1) * 16 + M) * 16 + lr] = sN[j];
  }
  __syncthreads();
  {
    int k = hf, M = lr;
    float s = 0.f;
#pragma unroll
    for (int q = 0; q < 16; ++q) s += sred[((w * 2 + k) * 16 + M) * 16 + q];
    scs[w * 32 + k * 16 + M] = s;
  }
  __syncthreads();
  if (hf == 0) {
    float s0 = scs[w * 32 + lr], s1 = scs[w * 32 + 16 + lr];
    float m = fmaxf(s0, s1);
    float e0 = expf(s0 - m), e1 = expf(s1 - m);
    al[w * 16 + lr] = e0 / (e0 + e1);
  }
  __syncthreads();
  for (int M = 0; M < 16; ++M) {
    int rg = tile * 16 + M;
    if (rg >= N) continue;
    float a0 = al[w * 16 + M], a1 = 1.0f - a0;
    int pv = (rg == 0) ? 1 : rg - 1;
    int nx = (rg == N - 1) ? N - 2 : rg + 1;
#pragma unroll
    for (int q = 0; q < 4; ++q) {
      int c = q * 32 + lane;
      float e = a0 * (float)Hbf[(size_t)pv * 128 + c] + a1 * (float)Hbf[(size_t)nx * 128 + c];
      Hcbf[(size_t)rg * 256 + 128 + c] = (__bf16)e;
      Hcbf[(size_t)rg * 256 + c] = Hbf[(size_t)rg * 128 + c];
    }
  }
}

// ---- k2b: P = tanh(Hc @ Wp + bp); ps = P @ templates.T  -> ps [N,10] ----
__global__ void k2b_patch(const __bf16* __restrict__ Hcbf, const __bf16* __restrict__ Wpt,
                          const float* __restrict__ bp, const float* __restrict__ templ,
                          float* __restrict__ ps, int N, int ntiles) {
  __shared__ float pL[2 * 16 * 256];
  int w = threadIdx.x >> 5, lane = threadIdx.x & 31;
  int hf = lane >> 4, lr = lane & 15;
  int tile = blockIdx.x * 2 + w; if (tile >= ntiles) tile = ntiles - 1;
  int row = tile * 16 + lr; if (row >= N) row = N - 1;
  f32x8 acc[16];
#pragma unroll
  for (int q = 0; q < 16; ++q) acc[q] = (f32x8)F32X8_ZERO;
  for (int kb = 0; kb < 256; kb += 32) {
    bf16x16 a = load_a_bf16(Hcbf, 256, row, kb, hf);
#pragma unroll
    for (int half = 0; half < 2; ++half) {     // two batches of 8 col-tiles
      bf16x16 bfr[8];
#pragma unroll
      for (int q = 0; q < 8; ++q)
        bfr[q] = load_b_bf16(Wpt, 256, (half * 8 + q) * 16 + lr, kb, hf);
#pragma unroll
      for (int q = 0; q < 8; ++q)
        acc[half * 8 + q] = wmma_bf16(a, bfr[q], acc[half * 8 + q]);
    }
  }
#pragma unroll
  for (int ct = 0; ct < 16; ++ct) {
    int col = ct * 16 + lr;
    float bb = bp[col];
#pragma unroll
    for (int j = 0; j < 8; ++j) {
      int M = j + hf * 8;
      pL[(w * 16 + M) * 256 + col] = tanhf(acc[ct][j] + bb);
    }
  }
  __syncthreads();
#pragma unroll
  for (int s = 0; s < 5; ++s) {
    int idx = lane * 5 + s;          // 0..159
    int M = idx / 10, t = idx % 10;
    float d = 0.f;
    for (int c = 0; c < 256; ++c) d += pL[(w * 16 + M) * 256 + c] * templ[t * 256 + c];
    int rg = tile * 16 + M;
    if (rg < N) ps[(size_t)rg * 10 + t] = d;
  }
}

// ---- k3/k4: max over N per template ----
__global__ void k3_pmax(const float* __restrict__ ps, float* __restrict__ pmax, int N) {
  __shared__ float red[256];
  float m[10];
#pragma unroll
  for (int t = 0; t < 10; ++t) m[t] = -3.4e38f;
  for (int n = blockIdx.x * 256 + threadIdx.x; n < N; n += gridDim.x * 256)
#pragma unroll
    for (int t = 0; t < 10; ++t) m[t] = fmaxf(m[t], ps[(size_t)n * 10 + t]);
  for (int t = 0; t < 10; ++t) {
    red[threadIdx.x] = m[t];
    __syncthreads();
    for (int s = 128; s > 0; s >>= 1) {
      if (threadIdx.x < s) red[threadIdx.x] = fmaxf(red[threadIdx.x], red[threadIdx.x + s]);
      __syncthreads();
    }
    if (threadIdx.x == 0) pmax[blockIdx.x * 10 + t] = red[0];
    __syncthreads();
  }
}

__global__ void k4_gmax(const float* __restrict__ pmax, float* __restrict__ gmax, int B) {
  int t = threadIdx.x;
  if (t < 10) {
    float m = -3.4e38f;
    for (int b = 0; b < B; ++b) m = fmaxf(m, pmax[b * 10 + t]);
    gmax[t] = m;
  }
}

// ---- k5: partial Z[t] and partial sum e*Hc -> pW [B2,10,256] ----
__global__ void k5_partial(const float* __restrict__ ps, const __bf16* __restrict__ Hc,
                           const float* __restrict__ gmax, float* __restrict__ pZ,
                           float* __restrict__ pW, int rpb, int N) {
  __shared__ float eL[1280];
  __shared__ float gm[10];
  int tid = threadIdx.x;
  if (tid < 10) gm[tid] = gmax[tid];
  float wacc[10];
#pragma unroll
  for (int t = 0; t < 10; ++t) wacc[t] = 0.f;
  float zacc = 0.f;
  int begin = blockIdx.x * rpb;
  int end = min(N, begin + rpb);
  __syncthreads();
  for (int cs = begin; cs < end; cs += 128) {
    int cnt = min(128, end - cs);
    for (int i = tid; i < cnt * 10; i += 256) {
      int r = i / 10, t = i - r * 10;
      eL[i] = expf(ps[(size_t)(cs + r) * 10 + t] - gm[t]);
    }
    __syncthreads();
    if (tid < 10) for (int r = 0; r < cnt; ++r) zacc += eL[r * 10 + tid];
    for (int r = 0; r < cnt; ++r) {
      float h = (float)Hc[(size_t)(cs + r) * 256 + tid];
      const float* er = &eL[r * 10];
#pragma unroll
      for (int t = 0; t < 10; ++t) wacc[t] += er[t] * h;
    }
    __syncthreads();
  }
  if (tid < 10) pZ[blockIdx.x * 10 + tid] = zacc;
#pragma unroll
  for (int t = 0; t < 10; ++t) pW[((size_t)blockIdx.x * 10 + t) * 256 + tid] = wacc[t];
}

// ---- k6: head: embs -> gammas -> M -> Y_prob/Y_hat, emit coef[t]=gamma_t/Z_t ----
__global__ void k6_final(const float* __restrict__ pZ, const float* __restrict__ pW,
                         const float* __restrict__ Wg1, const float* __restrict__ bg1,
                         const float* __restrict__ Wg2, const float* __restrict__ bg2,
                         const float* __restrict__ Wc, const float* __restrict__ bc,
                         float* __restrict__ coef, float* __restrict__ out, int B) {
  __shared__ float Zs[10], embs[2560], gL[1280], tss[10], gams[10], Ms[256], yp[2];
  int tid = threadIdx.x;
  if (tid < 10) {
    float s = 0.f;
    for (int b = 0; b < B; ++b) s += pZ[b * 10 + tid];
    Zs[tid] = s;
  }
  __syncthreads();
  for (int i = tid; i < 2560; i += 256) {
    int t = i >> 8, c = i & 255;
    float s = 0.f;
    for (int b = 0; b < B; ++b) s += pW[((size_t)b * 10 + t) * 256 + c];
    embs[i] = s / Zs[t];
  }
  __syncthreads();
  for (int i = tid; i < 1280; i += 256) {
    int t = i >> 7, d = i & 127;
    float s = bg1[d];
    for (int c = 0; c < 256; ++c) s += embs[t * 256 + c] * Wg1[c * 128 + d];
    gL[i] = tanhf(s);
  }
  __syncthreads();
  if (tid < 10) {
    float s = bg2[0];
    for (int d = 0; d < 128; ++d) s += gL[tid * 128 + d] * Wg2[d];
    tss[tid] = s;
  }
  __syncthreads();
  if (tid == 0) {
    float m = tss[0];
    for (int t = 1; t < 10; ++t) m = fmaxf(m, tss[t]);
    float se = 0.f;
    for (int t = 0; t < 10; ++t) { float e = expf(tss[t] - m); gams[t] = e; se += e; }
    for (int t = 0; t < 10; ++t) gams[t] /= se;
  }
  __syncthreads();
  {
    float s = 0.f;
#pragma unroll
    for (int t = 0; t < 10; ++t) s += gams[t] * embs[t * 256 + tid];
    Ms[tid] = s;
  }
  __syncthreads();
  if (tid < 2) {
    float s = bc[tid];
    for (int j = 0; j < 256; ++j) s += Ms[j] * Wc[j * 2 + tid];
    yp[tid] = 1.0f / (1.0f + expf(-s));
  }
  __syncthreads();
  if (tid == 0) {
    out[0] = yp[0];
    out[1] = yp[1];
    out[2] = (yp[1] > yp[0]) ? 1.0f : 0.0f;  // argmax
  }
  if (tid < 10) coef[tid] = gams[tid] / Zs[tid];
}

// ---- k7: A[n] = sum_t coef[t] * exp(ps[n,t]-gmax[t]) ----
__global__ void k7_A(const float* __restrict__ ps, const float* __restrict__ coef,
                     const float* __restrict__ gmax, float* __restrict__ outA, int N) {
  int n = blockIdx.x * 256 + threadIdx.x;
  if (n < N) {
    float s = 0.f;
#pragma unroll
    for (int t = 0; t < 10; ++t) s += coef[t] * expf(ps[(size_t)n * 10 + t] - gmax[t]);
    outA[n] = s;
  }
}

extern "C" void kernel_launch(void* const* d_in, const int* in_sizes, int n_in,
                              void* d_out, int out_size, void* d_ws, size_t ws_size,
                              hipStream_t stream) {
  (void)n_in; (void)out_size; (void)ws_size;
  const float* x         = (const float*)d_in[0];
  const float* W1        = (const float*)d_in[1];
  const float* b1        = (const float*)d_in[2];
  const float* Wn        = (const float*)d_in[3];
  const float* bn        = (const float*)d_in[4];
  const float* templates = (const float*)d_in[5];
  const float* Wp        = (const float*)d_in[6];
  const float* bp        = (const float*)d_in[7];
  const float* Wg1       = (const float*)d_in[8];
  const float* bg1       = (const float*)d_in[9];
  const float* Wg2       = (const float*)d_in[10];
  const float* bg2       = (const float*)d_in[11];
  const float* Wc        = (const float*)d_in[12];
  const float* bc        = (const float*)d_in[13];

  int N = in_sizes[0] / 512;
  int ntiles = (N + 15) / 16;

  char* ws = (char*)d_ws;
  size_t off = 0;
  auto take = [&](size_t bytes) -> char* {
    char* p = ws + off;
    off = (off + bytes + 255) & ~(size_t)255;
    return p;
  };
  __bf16* W1t  = (__bf16*)take((size_t)512 * 128 * 2);
  __bf16* Wnt  = (__bf16*)take((size_t)128 * 128 * 2);
  __bf16* Wpt  = (__bf16*)take((size_t)256 * 256 * 2);
  __bf16* Hbf  = (__bf16*)take((size_t)N * 128 * 2);
  __bf16* Hcbf = (__bf16*)take((size_t)N * 256 * 2);
  float*  ps   = (float*)take((size_t)N * 10 * 4);
  const int B1 = 250, B2 = 250;
  float* pmax = (float*)take((size_t)B1 * 10 * 4);
  float* gmax = (float*)take(10 * 4);
  float* pZ   = (float*)take((size_t)B2 * 10 * 4);
  float* pW   = (float*)take((size_t)B2 * 10 * 256 * 4);
  float* coef = (float*)take(10 * 4);

  float* out = (float*)d_out;
  int rpb = (N + B2 - 1) / B2;

  k0_convert<<<256, 256, 0, stream>>>(W1, Wn, Wp, W1t, Wnt, Wpt);
  k1_h<<<(ntiles + 7) / 8, 256, 0, stream>>>(x, W1t, b1, Hbf, N, ntiles);
  k2a_attn<<<(ntiles + 7) / 8, 256, 0, stream>>>(Hbf, Wnt, bn, Hcbf, N, ntiles);
  k2b_patch<<<(ntiles + 1) / 2, 64, 0, stream>>>(Hcbf, Wpt, bp, templates, ps, N, ntiles);
  k3_pmax<<<B1, 256, 0, stream>>>(ps, pmax, N);
  k4_gmax<<<1, 32, 0, stream>>>(pmax, gmax, B1);
  k5_partial<<<B2, 256, 0, stream>>>(ps, Hcbf, gmax, pZ, pW, rpb, N);
  k6_final<<<1, 256, 0, stream>>>(pZ, pW, Wg1, bg1, Wg2, bg2, Wc, bc, coef, out, B2);
  k7_A<<<(N + 255) / 256, 256, 0, stream>>>(ps, coef, gmax, out + 3, N);
}
